// MultiHeadAttention_22952305230212
// MI455X (gfx1250) — matrix operand order repaired and checked by reading, not compiled, not run on hardware
//
#include <hip/hip_runtime.h>
#include <hip/hip_bf16.h>

#define BATCH 4
#define SEQ   2048
#define D_MODEL 1024
#define NH    16
#define HD    64

typedef __attribute__((ext_vector_type(16))) __bf16 v16bf;
typedef __attribute__((ext_vector_type(8)))  float  v8f;

// ---- WMMA fragment loaders (CDNA5 16x16x32 bf16 layouts) -------------------
__device__ __forceinline__ v16bf load_a_frag(const __bf16* p, int ld) {
  const int lane = threadIdx.x & 31;
  const __bf16* row = p + (lane & 15) * ld + ((lane >> 4) << 3);
  v16bf a;
#pragma unroll
  for (int i = 0; i < 8; ++i) { a[i] = row[i]; a[i + 8] = row[i + 16]; }
  return a;
}

__device__ __forceinline__ v16bf load_b_frag(const __bf16* p, int ld) {
  const int lane = threadIdx.x & 31;
  const __bf16* col = p + (lane & 15) * ld + ((lane >> 4) << 3);
  v16bf b;
#pragma unroll
  for (int i = 0; i < 8; ++i) { b[i] = col[i]; b[i + 8] = col[i + 16]; }
  return b;
}

__device__ __forceinline__ v8f wmma_bf16(v16bf a, v16bf b, v8f c) {
  return __builtin_amdgcn_wmma_f32_16x16x32_bf16(false, a, false, b, (short)0, c,
                                                 false, false);
}

// ---- CDNA5 async global->LDS copy (ASYNCcnt-tracked) -----------------------
// Generic pointer to __shared__ has the LDS byte offset in its low 32 bits.
__device__ __forceinline__ unsigned lds_off(const void* p) {
  return (unsigned)(uintptr_t)p;
}
__device__ __forceinline__ void async_ld16(unsigned loff,
                                           const void* gbase, unsigned goff) {
  asm volatile("global_load_async_to_lds_b128 %0, %1, %2 offset:0"
               :: "v"(loff), "v"(goff),
                  "s"((unsigned long long)(uintptr_t)gbase)
               : "memory");
}

// ---- Kernel 1: fused QKV projection (fp32 in -> bf16 WMMA -> bf16 out) -----
__global__ __launch_bounds__(256) void proj_qkv_kernel(
    const float* __restrict__ Xq, const float* __restrict__ Xk,
    const float* __restrict__ Xv, const float* __restrict__ Wq,
    const float* __restrict__ Wk, const float* __restrict__ Wv,
    const float* __restrict__ bq, const float* __restrict__ bk,
    const float* __restrict__ bv, __bf16* __restrict__ qo,
    __bf16* __restrict__ ko, __bf16* __restrict__ vto) {
  __shared__ __bf16 As[128][40];
  __shared__ __bf16 Bs[128][40];
  const int z = blockIdx.z;
  const float* X = (z == 0) ? Xq : (z == 1) ? Xk : Xv;
  const float* W = (z == 0) ? Wq : (z == 1) ? Wk : Wv;
  const float* bias = (z == 0) ? bq : (z == 1) ? bk : bv;
  const int n0 = blockIdx.x * 128;
  const int m0 = blockIdx.y * 128;
  const int t = threadIdx.x, wave = t >> 5, lane = t & 31;
  const int wm = wave >> 2, wn = wave & 3;
  v8f acc[4][2] = {};
  for (int k0 = 0; k0 < D_MODEL; k0 += 32) {
    if (k0 + 32 < D_MODEL) {  // prefetch next K-slab (global_prefetch_b8)
      __builtin_prefetch(&X[(size_t)(m0 + (t >> 1)) * D_MODEL + k0 + 32 + (t & 1) * 16], 0, 1);
      __builtin_prefetch(&W[(size_t)(k0 + 32 + (t >> 3)) * D_MODEL + n0 + (t & 7) * 16], 0, 1);
    }
#pragma unroll
    for (int i = 0; i < 16; ++i) {
      int e = t + i * 256, row = e >> 5, kk = e & 31;
      As[row][kk] = (__bf16)X[(size_t)(m0 + row) * D_MODEL + k0 + kk];
    }
#pragma unroll
    for (int i = 0; i < 16; ++i) {
      int e = t + i * 256, kk = e >> 7, col = e & 127;
      Bs[col][kk] = (__bf16)W[(size_t)(k0 + kk) * D_MODEL + n0 + col];
    }
    __syncthreads();
#pragma unroll
    for (int ms = 0; ms < 4; ++ms) {
      v16bf a = load_a_frag(&As[wm * 64 + ms * 16][0], 40);
#pragma unroll
      for (int ns = 0; ns < 2; ++ns) {
        v16bf bfr = load_b_frag(&Bs[wn * 32 + ns * 16][0], 40);
        acc[ms][ns] = wmma_bf16(a, bfr, acc[ms][ns]);
      }
    }
    __syncthreads();
  }
  const int rowoff = (lane >> 4) << 3, coll = lane & 15;
#pragma unroll
  for (int ms = 0; ms < 4; ++ms)
#pragma unroll
    for (int ns = 0; ns < 2; ++ns) {
      int n = n0 + wn * 32 + ns * 16 + coll;
      float bb = bias[n];
      int h = n >> 6, d = n & 63;
#pragma unroll
      for (int r = 0; r < 8; ++r) {
        int m = m0 + wm * 64 + ms * 16 + rowoff + r;
        int b = m >> 11, s = m & (SEQ - 1);
        float val = acc[ms][ns][r] + bb;
        if (z == 2)
          vto[(((size_t)b * NH + h) * HD + d) * SEQ + s] = (__bf16)val;
        else {
          size_t idx = (((size_t)b * NH + h) * SEQ + s) * HD + d;
          if (z == 0) qo[idx] = (__bf16)val; else ko[idx] = (__bf16)val;
        }
      }
    }
}

// ---- Kernel 2: flash attention with async-LDS staged K/V/mask --------------
// grid: (S/64=32, H=16, B=4)  block: 128 (4 waves, 16 q-rows each)
__global__ __launch_bounds__(128) void attn_kernel(
    const __bf16* __restrict__ qw, const __bf16* __restrict__ kw,
    const __bf16* __restrict__ vt, const unsigned char* __restrict__ mask,
    __bf16* __restrict__ ctx, float* __restrict__ attn) {
  __shared__ __bf16 Ks[2][64][72];        // [key][d], d contiguous
  __shared__ __bf16 Vs[2][64][72];        // [d][key], key contiguous
  __shared__ unsigned char Ms[2][64][64]; // [q_local][key_local]
  __shared__ __bf16 plds[4][16][72];      // per-wave P tile
  const int t = threadIdx.x, wave = t >> 5, lane = t & 31;
  const int h = blockIdx.y, b = blockIdx.z;
  const int bh = b * NH + h;
  const int qblk = blockIdx.x * 64;
  const int q0 = qblk + wave * 16;
  const int coll = lane & 15;
  const int rowoff = (lane >> 4) << 3;

  const __bf16* kbase_g = kw + (size_t)bh * SEQ * HD;
  const __bf16* vbase_g = vt + (size_t)bh * HD * SEQ;
  const unsigned char* mbase_g = mask + ((size_t)b * SEQ + qblk) * SEQ;

  // issue async copies for one key tile: K(4) + V(4) + M(2) chunks per thread
  auto issue_kvm = [&](int kt, int bufi) {
#pragma unroll
    for (int i = 0; i < 4; ++i) {  // K tile: 64 keys x 128B
      int ch = t + i * 128, row = ch >> 3, c = ch & 7;
      async_ld16(lds_off(&Ks[bufi][row][0]) + c * 16, kbase_g,
                 (unsigned)((kt + row) * HD * 2 + c * 16));
    }
#pragma unroll
    for (int i = 0; i < 4; ++i) {  // V tile: 64 d x 128B
      int ch = t + i * 128, row = ch >> 3, c = ch & 7;
      async_ld16(lds_off(&Vs[bufi][row][0]) + c * 16, vbase_g,
                 (unsigned)((row * SEQ + kt) * 2 + c * 16));
    }
#pragma unroll
    for (int i = 0; i < 2; ++i) {  // mask tile: 64 q x 64B
      int ch = t + i * 128, row = ch >> 2, c = ch & 3;
      async_ld16(lds_off(&Ms[bufi][row][0]) + c * 16, mbase_g,
                 (unsigned)(row * SEQ + kt + c * 16));
    }
  };
  auto issue_km = [&](int kt, int bufi) {  // pass 2: K(4) + M(2)
#pragma unroll
    for (int i = 0; i < 4; ++i) {
      int ch = t + i * 128, row = ch >> 3, c = ch & 7;
      async_ld16(lds_off(&Ks[bufi][row][0]) + c * 16, kbase_g,
                 (unsigned)((kt + row) * HD * 2 + c * 16));
    }
#pragma unroll
    for (int i = 0; i < 2; ++i) {
      int ch = t + i * 128, row = ch >> 2, c = ch & 3;
      async_ld16(lds_off(&Ms[bufi][row][0]) + c * 16, mbase_g,
                 (unsigned)(row * SEQ + kt + c * 16));
    }
  };

  const __bf16* qbase = qw + ((size_t)bh * SEQ + q0) * HD;
  v16bf qf0 = load_a_frag(qbase, HD);
  v16bf qf1 = load_a_frag(qbase + 32, HD);

  float mrow[8], lrow[8];
  v8f oacc[4] = {};
#pragma unroll
  for (int r = 0; r < 8; ++r) { mrow[r] = -3.0e38f; lrow[r] = 0.f; }
  v8f zeroc = {};

  issue_kvm(0, 0);
  for (int n = 0; n < SEQ / 64; ++n) {
    const int buf = n & 1;
    if (n + 1 < SEQ / 64) {
      issue_kvm((n + 1) * 64, (n + 1) & 1);
      asm volatile("s_wait_asynccnt 0xa" ::: "memory");  // tile n complete
    } else {
      asm volatile("s_wait_asynccnt 0x0" ::: "memory");
    }
    __syncthreads();

    v8f p[4];
    float tmax[8], tsum[8];
#pragma unroll
    for (int r = 0; r < 8; ++r) { tmax[r] = -3.0e38f; tsum[r] = 0.f; }
#pragma unroll
    for (int sub = 0; sub < 4; ++sub) {
      v16bf b0 = load_b_frag(&Ks[buf][sub * 16][0], 72);
      v16bf b1 = load_b_frag(&Ks[buf][sub * 16][0] + 32, 72);
      v8f s = wmma_bf16(qf0, b0, zeroc);
      s = wmma_bf16(qf1, b1, s);
#pragma unroll
      for (int r = 0; r < 8; ++r) {
        float sv = s[r] * 0.125f;  // 1/sqrt(64)
        if (Ms[buf][wave * 16 + rowoff + r][sub * 16 + coll]) sv = -1.0e9f;
        s[r] = sv;
        tmax[r] = fmaxf(tmax[r], sv);
      }
      p[sub] = s;
    }
#pragma unroll
    for (int r = 0; r < 8; ++r) {
      float v = tmax[r];
      v = fmaxf(v, __shfl_xor(v, 1)); v = fmaxf(v, __shfl_xor(v, 2));
      v = fmaxf(v, __shfl_xor(v, 4)); v = fmaxf(v, __shfl_xor(v, 8));
      tmax[r] = v;
    }
    float alpha[8];
#pragma unroll
    for (int r = 0; r < 8; ++r) {
      float mnew = fmaxf(mrow[r], tmax[r]);
      alpha[r] = __expf(mrow[r] - mnew);
      mrow[r] = mnew;
    }
#pragma unroll
    for (int sub = 0; sub < 4; ++sub)
#pragma unroll
      for (int r = 0; r < 8; ++r) {
        float e = __expf(p[sub][r] - mrow[r]);
        tsum[r] += e;
        plds[wave][rowoff + r][sub * 16 + coll] = (__bf16)e;
      }
#pragma unroll
    for (int r = 0; r < 8; ++r) {
      float v = tsum[r];
      v += __shfl_xor(v, 1); v += __shfl_xor(v, 2);
      v += __shfl_xor(v, 4); v += __shfl_xor(v, 8);
      lrow[r] = lrow[r] * alpha[r] + v;
    }
#pragma unroll
    for (int dsub = 0; dsub < 4; ++dsub)
#pragma unroll
      for (int r = 0; r < 8; ++r) oacc[dsub][r] *= alpha[r];

    v16bf pa0 = load_a_frag(&plds[wave][0][0], 72);
    v16bf pa1 = load_a_frag(&plds[wave][0][0] + 32, 72);
#pragma unroll
    for (int dsub = 0; dsub < 4; ++dsub) {
      v16bf v0 = load_b_frag(&Vs[buf][dsub * 16][0], 72);
      v16bf v1 = load_b_frag(&Vs[buf][dsub * 16][0] + 32, 72);
      oacc[dsub] = wmma_bf16(pa0, v0, oacc[dsub]);
      oacc[dsub] = wmma_bf16(pa1, v1, oacc[dsub]);
    }
    __syncthreads();  // buffer reusable for tile n+2
  }
  // context out (bf16, [B*S, H*64] row-major)
#pragma unroll
  for (int dsub = 0; dsub < 4; ++dsub)
#pragma unroll
    for (int r = 0; r < 8; ++r) {
      size_t row = (size_t)b * SEQ + q0 + rowoff + r;
      ctx[row * D_MODEL + h * HD + dsub * 16 + coll] =
          (__bf16)(oacc[dsub][r] / lrow[r]);
    }
  // ---- pass 2: recompute scores, stream normalized probabilities ----------
  float linv[8];
#pragma unroll
  for (int r = 0; r < 8; ++r) linv[r] = 1.0f / lrow[r];
  issue_km(0, 0);
  for (int n = 0; n < SEQ / 64; ++n) {
    const int buf = n & 1;
    const int kt = n * 64;
    if (n + 1 < SEQ / 64) {
      issue_km((n + 1) * 64, (n + 1) & 1);
      asm volatile("s_wait_asynccnt 0x6" ::: "memory");
    } else {
      asm volatile("s_wait_asynccnt 0x0" ::: "memory");
    }
    __syncthreads();
#pragma unroll
    for (int sub = 0; sub < 4; ++sub) {
      v16bf b0 = load_b_frag(&Ks[buf][sub * 16][0], 72);
      v16bf b1 = load_b_frag(&Ks[buf][sub * 16][0] + 32, 72);
      v8f s = wmma_bf16(qf0, b0, zeroc);
      s = wmma_bf16(qf1, b1, s);
      int key = kt + sub * 16 + coll;
#pragma unroll
      for (int r = 0; r < 8; ++r) {
        float sv = s[r] * 0.125f;
        if (Ms[buf][wave * 16 + rowoff + r][sub * 16 + coll]) sv = -1.0e9f;
        attn[((size_t)bh * SEQ + q0 + rowoff + r) * SEQ + key] =
            __expf(sv - mrow[r]) * linv[r];
      }
    }
    __syncthreads();
  }
}

// ---- Kernel 3: output projection (bf16 ctx @ fp32 Wo + bo -> fp32) ---------
__global__ __launch_bounds__(256) void out_proj_kernel(
    const __bf16* __restrict__ A, const float* __restrict__ W,
    const float* __restrict__ bias, float* __restrict__ out) {
  __shared__ __bf16 As[128][40];
  __shared__ __bf16 Bs[128][40];
  const int n0 = blockIdx.x * 128, m0 = blockIdx.y * 128;
  const int t = threadIdx.x, wave = t >> 5, lane = t & 31;
  const int wm = wave >> 2, wn = wave & 3;
  v8f acc[4][2] = {};
  for (int k0 = 0; k0 < D_MODEL; k0 += 32) {
    if (k0 + 32 < D_MODEL)
      __builtin_prefetch(&W[(size_t)(k0 + 32 + (t >> 3)) * D_MODEL + n0 + (t & 7) * 16], 0, 1);
#pragma unroll
    for (int i = 0; i < 16; ++i) {
      int e = t + i * 256, row = e >> 5, kk = e & 31;
      As[row][kk] = A[(size_t)(m0 + row) * D_MODEL + k0 + kk];
    }
#pragma unroll
    for (int i = 0; i < 16; ++i) {
      int e = t + i * 256, kk = e >> 7, col = e & 127;
      Bs[col][kk] = (__bf16)W[(size_t)(k0 + kk) * D_MODEL + n0 + col];
    }
    __syncthreads();
#pragma unroll
    for (int ms = 0; ms < 4; ++ms) {
      v16bf a = load_a_frag(&As[wm * 64 + ms * 16][0], 40);
#pragma unroll
      for (int ns = 0; ns < 2; ++ns) {
        v16bf bfr = load_b_frag(&Bs[wn * 32 + ns * 16][0], 40);
        acc[ms][ns] = wmma_bf16(a, bfr, acc[ms][ns]);
      }
    }
    __syncthreads();
  }
  const int rowoff = (lane >> 4) << 3, coll = lane & 15;
#pragma unroll
  for (int ms = 0; ms < 4; ++ms)
#pragma unroll
    for (int ns = 0; ns < 2; ++ns) {
      int n = n0 + wn * 32 + ns * 16 + coll;
      float bb = bias[n];
#pragma unroll
      for (int r = 0; r < 8; ++r) {
        int m = m0 + wm * 64 + ms * 16 + rowoff + r;
        out[(size_t)m * D_MODEL + n] = acc[ms][ns][r] + bb;
      }
    }
}

// ---- Kernel 4: residual + layernorm ---------------------------------------
__global__ __launch_bounds__(256) void ln_kernel(
    const float* __restrict__ y, const float* __restrict__ resid,
    const float* __restrict__ gamma, const float* __restrict__ beta,
    float* __restrict__ out) {
  const int row = blockIdx.x;
  const float* x0 = y + (size_t)row * D_MODEL;
  const float* r0 = resid + (size_t)row * D_MODEL;
  float xv[4], s1 = 0.f, s2 = 0.f;
#pragma unroll
  for (int i = 0; i < 4; ++i) {
    int j = threadIdx.x + i * 256;
    float v = x0[j] + r0[j];
    xv[i] = v; s1 += v; s2 += v * v;
  }
#pragma unroll
  for (int off = 16; off >= 1; off >>= 1) {
    s1 += __shfl_xor(s1, off); s2 += __shfl_xor(s2, off);
  }
  __shared__ float red1[8], red2[8];
  int wave = threadIdx.x >> 5, lane = threadIdx.x & 31;
  if (lane == 0) { red1[wave] = s1; red2[wave] = s2; }
  __syncthreads();
  s1 = 0.f; s2 = 0.f;
#pragma unroll
  for (int w = 0; w < 8; ++w) { s1 += red1[w]; s2 += red2[w]; }
  float mu = s1 * (1.0f / D_MODEL);
  float var = s2 * (1.0f / D_MODEL) - mu * mu;
  float inv = rsqrtf(var + 1e-5f);
#pragma unroll
  for (int i = 0; i < 4; ++i) {
    int j = threadIdx.x + i * 256;
    out[(size_t)row * D_MODEL + j] = (xv[i] - mu) * inv * gamma[j] + beta[j];
  }
}

extern "C" void kernel_launch(void* const* d_in, const int* in_sizes, int n_in,
                              void* d_out, int out_size, void* d_ws,
                              size_t ws_size, hipStream_t stream) {
  const float* query = (const float*)d_in[0];
  const float* key_ = (const float*)d_in[1];
  const float* value = (const float*)d_in[2];
  const unsigned char* mask = (const unsigned char*)d_in[3];
  const float* Wq = (const float*)d_in[4];
  const float* bq = (const float*)d_in[5];
  const float* Wk = (const float*)d_in[6];
  const float* bk = (const float*)d_in[7];
  const float* Wv = (const float*)d_in[8];
  const float* bv = (const float*)d_in[9];
  const float* Wo = (const float*)d_in[10];
  const float* bo = (const float*)d_in[11];
  const float* gamma = (const float*)d_in[12];
  const float* beta = (const float*)d_in[13];

  float* out_ln = (float*)d_out;
  float* out_attn = out_ln + (size_t)BATCH * SEQ * D_MODEL;

  char* ws = (char*)d_ws;
  __bf16* q_ws = (__bf16*)ws;                           // 16 MiB
  __bf16* k_ws = (__bf16*)(ws + ((size_t)16 << 20));    // 16 MiB
  __bf16* vt_ws = (__bf16*)(ws + ((size_t)32 << 20));   // 16 MiB
  __bf16* ctx_ws = (__bf16*)(ws + ((size_t)48 << 20));  // 16 MiB
  float* tmp_ws = (float*)(ws + ((size_t)64 << 20));    // 32 MiB

  proj_qkv_kernel<<<dim3(8, 64, 3), 256, 0, stream>>>(
      query, key_, value, Wq, Wk, Wv, bq, bk, bv, q_ws, k_ws, vt_ws);
  attn_kernel<<<dim3(SEQ / 64, NH, BATCH), 128, 0, stream>>>(
      q_ws, k_ws, vt_ws, mask, ctx_ws, out_attn);
  out_proj_kernel<<<dim3(8, 64), 256, 0, stream>>>(ctx_ws, Wo, bo, tmp_ws);
  ln_kernel<<<BATCH * SEQ, 256, 0, stream>>>(tmp_ws, query, gamma, beta, out_ln);
}
